// BeamGNN_71932112273657
// MI455X (gfx1250) — matrix operand order repaired and checked by reading, not compiled, not run on hardware
//
#include <hip/hip_runtime.h>
#include <hip/hip_bf16.h>
#include <math.h>

typedef __attribute__((ext_vector_type(16))) _Float16 v16h;
typedef __attribute__((ext_vector_type(8)))  float    v8f;

#define HID 32

// ---------------------------------------------------------------------------
// Degree / normalization
// ---------------------------------------------------------------------------
__global__ void k_deg_init(float* __restrict__ deg, int n) {
    int i = blockIdx.x * blockDim.x + threadIdx.x;
    if (i < n) deg[i] = 1.0f;               // self-loop contributes 1 to in-degree
}

__global__ void k_deg_count(const int* __restrict__ dst, float* __restrict__ deg, int nE) {
    int e = blockIdx.x * blockDim.x + threadIdx.x;
    if (e < nE) unsafeAtomicAdd(&deg[dst[e]], 1.0f);   // global_atomic_add_f32
}

__global__ void k_deg_to_dis(float* __restrict__ dis, int n) {
    int i = blockIdx.x * blockDim.x + threadIdx.x;
    if (i < n) dis[i] = rsqrtf(dis[i]);     // deg >= 1 always (self-loops)
}

// Per-edge normalization weight, reused by all 4 scatter passes.
__global__ void k_edge_w(const int* __restrict__ src, const int* __restrict__ dst,
                         const float* __restrict__ dis, float* __restrict__ w, int nE) {
    int e = blockIdx.x * blockDim.x + threadIdx.x;
    if (e < nE) w[e] = dis[src[e]] * dis[dst[e]];
}

// ---------------------------------------------------------------------------
// WMMA dense matmul: h[N,32] = act(x)[N,F_IN] @ W[F_IN,32]  (f16 in, f32 accum)
// One wave per 16-row tile; two 16x16x32 WMMAs cover the 32 output columns.
// TANH_IN applies tanh to the input features on load (fuses prev activation).
// ---------------------------------------------------------------------------
template <int F_IN, bool TANH_IN>
__global__ void k_matmul_wmma(const float* __restrict__ x,
                              const float* __restrict__ W,   // row-major [F_IN][32]
                              float* __restrict__ h, int n_nodes) {
    const int lane = threadIdx.x & 31;
    const int wave = threadIdx.x >> 5;
    const int tile = blockIdx.x * (blockDim.x >> 5) + wave;
    const int row0 = tile * 16;
    if (row0 >= n_nodes) return;            // wave-uniform exit: EXEC stays all-1s

    // ---- B fragments: W columns 0-15 (b0) and 16-31 (b1) -------------------
    // Layout: lane n = col; halves 0-7 hold K = kb16+0..7, halves 8-15 hold
    // K = 16+kb16+0..7 (kb16 = 0 for lanes 0-15, 8 for lanes 16-31).
    const int nlo  = lane & 15;
    const int kb16 = (lane >> 4) * 8;
    v16h b0, b1;
#pragma unroll
    for (int hh = 0; hh < 16; ++hh) {
        const int k = (hh < 8) ? (kb16 + hh) : (16 + kb16 + (hh - 8));
        float w0 = 0.f, w1 = 0.f;
        if (k < F_IN) {
            w0 = W[k * HID + nlo];
            w1 = W[k * HID + nlo + 16];
        }
        b0[hh] = (_Float16)w0;
        b1[hh] = (_Float16)w1;
    }

    // ---- A fragment: 16x32 tile of act(x) ----------------------------------
    // Layout: lane m = row; halves 0-7 hold K = kb8+0..7, halves 8-15 hold
    // K = 16+kb8+0..7 (kb8 = 0 for lanes 0-15, 8 for lanes 16-31).
    const int m   = lane & 15;
    const int kb8 = (lane >> 4) * 8;
    const int row = row0 + m;
    const bool full_tile = (row0 + 16 <= n_nodes);   // wave-uniform

    v16h a;
#pragma unroll
    for (int hh = 0; hh < 16; ++hh) a[hh] = (_Float16)0.f;

    if (full_tile && F_IN == HID) {
        // Two contiguous 8-float runs -> four b128 loads per lane.
        const float4* xr = (const float4*)(x + (size_t)row * HID);
        float4 p[4];
        p[0] = xr[(kb8 >> 2) + 0];
        p[1] = xr[(kb8 >> 2) + 1];
        p[2] = xr[4 + (kb8 >> 2) + 0];
        p[3] = xr[4 + (kb8 >> 2) + 1];
        float v[16] = { p[0].x, p[0].y, p[0].z, p[0].w,  p[1].x, p[1].y, p[1].z, p[1].w,
                        p[2].x, p[2].y, p[2].z, p[2].w,  p[3].x, p[3].y, p[3].z, p[3].w };
#pragma unroll
        for (int hh = 0; hh < 16; ++hh) {
            float t = v[hh];
            if (TANH_IN) t = tanhf(t);
            a[hh] = (_Float16)t;
        }
    } else {
#pragma unroll
        for (int hh = 0; hh < 16; ++hh) {
            const int k = (hh < 8) ? (kb8 + hh) : (16 + kb8 + (hh - 8));
            if (k < F_IN && row < n_nodes) {
                float t = x[(size_t)row * F_IN + k];
                if (TANH_IN) t = tanhf(t);
                a[hh] = (_Float16)t;
            }
        }
    }

    v8f c0 = {};
    v8f c1 = {};
    c0 = __builtin_amdgcn_wmma_f32_16x16x32_f16(false, a, false, b0, (short)0, c0, false, false);
    c1 = __builtin_amdgcn_wmma_f32_16x16x32_f16(false, a, false, b1, (short)0, c1, false, false);

    // ---- D store: lane n = col; vgpr r -> M = 8*(lane>=16) + r -------------
    const int nn = lane & 15;
    const int mb = (lane >> 4) * 8;
    if (full_tile) {
        float* hp = h + (size_t)(row0 + mb) * HID + nn;
#pragma unroll
        for (int r = 0; r < 8; ++r) {
            hp[(size_t)r * HID]      = c0[r];
            hp[(size_t)r * HID + 16] = c1[r];
        }
    } else {
#pragma unroll
        for (int r = 0; r < 8; ++r) {
            const int rr = row0 + mb + r;
            if (rr < n_nodes) {
                h[(size_t)rr * HID + nn]      = c0[r];
                h[(size_t)rr * HID + nn + 16] = c1[r];
            }
        }
    }
}

// ---------------------------------------------------------------------------
// acc[i][:] = h[i][:] * dis[i]^2 + b[:]   (self-loop term + bias, float4 wide)
// ---------------------------------------------------------------------------
__global__ void k_selfinit(const float* __restrict__ h, const float* __restrict__ dis,
                           const float* __restrict__ b, float* __restrict__ acc, int n) {
    int idx = blockIdx.x * blockDim.x + threadIdx.x;   // n * 8 threads
    if (idx >= n * 8) return;
    const int i = idx >> 3, part = idx & 7;
    float w = dis[i]; w *= w;
    const float4 hv = ((const float4*)(h + (size_t)i * HID))[part];
    const float4 bb = ((const float4*)b)[part];
    float4 r;
    r.x = hv.x * w + bb.x; r.y = hv.y * w + bb.y;
    r.z = hv.z * w + bb.z; r.w = hv.w * w + bb.w;
    ((float4*)(acc + (size_t)i * HID))[part] = r;
}

// ---------------------------------------------------------------------------
// Edge scatter: acc[dst] += h[src] * w_e
// 8 lanes per edge, float4 gather + 4 hardware f32 atomics each.
// ---------------------------------------------------------------------------
__global__ void k_edge_scatter32(const int* __restrict__ src, const int* __restrict__ dst,
                                 const float* __restrict__ ew, const float* __restrict__ h,
                                 float* __restrict__ acc, int nE) {
    int idx = blockIdx.x * blockDim.x + threadIdx.x;   // nE * 8 threads
    if (idx >= nE * 8) return;
    const int e = idx >> 3, part = idx & 7;
    const int s = src[e], d = dst[e];
    const float we = ew[e];
    const float4 hv = ((const float4*)(h + (size_t)s * HID))[part];
    float* ap = acc + (size_t)d * HID + part * 4;
    unsafeAtomicAdd(ap + 0, hv.x * we);
    unsafeAtomicAdd(ap + 1, hv.y * we);
    unsafeAtomicAdd(ap + 2, hv.z * we);
    unsafeAtomicAdd(ap + 3, hv.w * we);
}

// ---------------------------------------------------------------------------
// Output layer: h4[i] = tanh(feat[i,:])·W4 ; out[i] = h4[i]*dis[i]^2 + b4
// ---------------------------------------------------------------------------
__global__ void k_out_dot(const float* __restrict__ feat, const float* __restrict__ W4,
                          const float* __restrict__ b4, const float* __restrict__ dis,
                          float* __restrict__ h4, float* __restrict__ out, int n) {
    int i = blockIdx.x * blockDim.x + threadIdx.x;
    if (i >= n) return;
    const float4* f = (const float4*)(feat + (size_t)i * HID);
    float s = 0.f;
#pragma unroll
    for (int q = 0; q < 8; ++q) {
        const float4 v = f[q];
        s += tanhf(v.x) * W4[q * 4 + 0] + tanhf(v.y) * W4[q * 4 + 1]
           + tanhf(v.z) * W4[q * 4 + 2] + tanhf(v.w) * W4[q * 4 + 3];
    }
    h4[i] = s;
    const float di = dis[i];
    out[i] = s * di * di + b4[0];
}

__global__ void k_edge_scatter1(const int* __restrict__ src, const int* __restrict__ dst,
                                const float* __restrict__ ew, const float* __restrict__ h4,
                                float* __restrict__ out, int nE) {
    int e = blockIdx.x * blockDim.x + threadIdx.x;
    if (e < nE) unsafeAtomicAdd(&out[dst[e]], h4[src[e]] * ew[e]);
}

// ---------------------------------------------------------------------------
// Launch
// ---------------------------------------------------------------------------
extern "C" void kernel_launch(void* const* d_in, const int* in_sizes, int n_in,
                              void* d_out, int out_size, void* d_ws, size_t ws_size,
                              hipStream_t stream) {
    const float* x  = (const float*)d_in[0];
    const int*   ei = (const int*)d_in[1];
    const float* W1 = (const float*)d_in[2];  const float* b1 = (const float*)d_in[3];
    const float* W2 = (const float*)d_in[4];  const float* b2 = (const float*)d_in[5];
    const float* W3 = (const float*)d_in[6];  const float* b3 = (const float*)d_in[7];
    const float* W4 = (const float*)d_in[8];  const float* b4 = (const float*)d_in[9];

    const int N = in_sizes[0] / 3;
    const int E = in_sizes[1] / 2;
    const int* src = ei;
    const int* dst = ei + E;

    // Workspace: dis[N] | ew[E] | fX[N*32] (features / accumulator) | fY[N*32] (h / h4)
    float* dis = (float*)d_ws;
    float* ew  = dis + (size_t)N;
    float* fX  = ew + (size_t)E;
    float* fY  = fX + (size_t)N * HID;
    float* out = (float*)d_out;

    const int T = 256;
    const int gN  = (N + T - 1) / T;
    const int gE  = (E + T - 1) / T;
    const int gN8 = (N * 8 + T - 1) / T;
    const int gE8 = (E * 8 + T - 1) / T;
    const int tiles = (N + 15) / 16;
    const int gMM = (tiles + 3) / 4;         // 4 waves (128 threads) per block

    // Normalization
    k_deg_init  <<<gN, T, 0, stream>>>(dis, N);
    k_deg_count <<<gE, T, 0, stream>>>(dst, dis, E);
    k_deg_to_dis<<<gN, T, 0, stream>>>(dis, N);
    k_edge_w    <<<gE, T, 0, stream>>>(src, dst, dis, ew, E);

    // Layer 1: x[N,3] -> fX (pre-activation; tanh fused into next consumer)
    k_matmul_wmma<3, false><<<gMM, 128, 0, stream>>>(x, W1, fY, N);
    k_selfinit      <<<gN8, T, 0, stream>>>(fY, dis, b1, fX, N);
    k_edge_scatter32<<<gE8, T, 0, stream>>>(src, dst, ew, fY, fX, E);

    // Layer 2
    k_matmul_wmma<HID, true><<<gMM, 128, 0, stream>>>(fX, W2, fY, N);
    k_selfinit      <<<gN8, T, 0, stream>>>(fY, dis, b2, fX, N);
    k_edge_scatter32<<<gE8, T, 0, stream>>>(src, dst, ew, fY, fX, E);

    // Layer 3
    k_matmul_wmma<HID, true><<<gMM, 128, 0, stream>>>(fX, W3, fY, N);
    k_selfinit      <<<gN8, T, 0, stream>>>(fY, dis, b3, fX, N);
    k_edge_scatter32<<<gE8, T, 0, stream>>>(src, dst, ew, fY, fX, E);

    // Output layer (tanh of layer-3 features fused; bias + self-loop fused)
    k_out_dot       <<<gN, T, 0, stream>>>(fX, W4, b4, dis, fY, out, N);
    k_edge_scatter1 <<<gE, T, 0, stream>>>(src, dst, ew, fY, out, E);
}
